// SemanticVectorQuantizer_55989193671484
// MI455X (gfx1250) — matrix operand order repaired and checked by reading, not compiled, not run on hardware
//
#include <hip/hip_runtime.h>
#include <hip/hip_bf16.h>

// ---------------------------------------------------------------------------
// SemanticVectorQuantizer forward for MI455X (gfx1250, wave32, bf16 WMMA,
// async global->LDS staging for the codebook-search GEMM)
//   B=128 N=64 D=2048 M=8 K=4096 C=256 BLK=256, COMMIT=0.25
// ---------------------------------------------------------------------------

#define B_    128
#define N_    64
#define D_    2048
#define M_    8
#define K_    4096
#define C_    256
#define BLK_  256
#define BN_   (B_*N_)      // 8192 rows

typedef __bf16 bf16x16 __attribute__((ext_vector_type(16)));
typedef float  f32x8   __attribute__((ext_vector_type(8)));

union Frag16  { bf16x16 v; unsigned u[8]; uint4 q[2]; };
union FragAcc { f32x8 v;   float   f[8]; };

__device__ __forceinline__ unsigned packh(__bf16 a, __bf16 b) {
  union { __bf16 h[2]; unsigned u; } x; x.h[0] = a; x.h[1] = b; return x.u;
}

__device__ __forceinline__ f32x8 wmma_bf16(const Frag16& a, const Frag16& b, f32x8 c) {
  // v_wmma_f32_16x16x32_bf16: (neg_a, A, neg_b, B, c_mod, C, reuse_a, reuse_b)
  return __builtin_amdgcn_wmma_f32_16x16x32_bf16(false, a.v, false, b.v,
                                                 (short)0, c, false, false);
}

// Async global->LDS copy of 16 bytes (GLOBAL_LOAD_ASYNC_TO_LDS_B128, ASYNCcnt).
// VDST VGPR carries the LDS byte offset; VADDR the 64-bit global address.
__device__ __forceinline__ void async_copy_b128(unsigned lds_off, const void* gaddr) {
  asm volatile("global_load_async_to_lds_b128 %0, %1, off"
               :: "v"(lds_off), "v"(gaddr)
               : "memory");
}
__device__ __forceinline__ void wait_asynccnt0() {
  asm volatile("s_wait_asynccnt 0x0" ::: "memory");
}

// ---------------------------------------------------------------------------
// Kernel 0: zero the scalar-loss accumulator
// ---------------------------------------------------------------------------
__global__ void zero_loss_kernel(float* loss) {
  if (threadIdx.x == 0) *loss = 0.0f;
}

// ---------------------------------------------------------------------------
// Kernel 1: codebooks -> bf16 copy + f32 squared norms.  One wave per (m,k).
// ---------------------------------------------------------------------------
__global__ __launch_bounds__(256) void prep_cb_kernel(const float* __restrict__ cb,
                                                      __bf16* __restrict__ cbb,
                                                      float* __restrict__ cnorm) {
  int w    = (blockIdx.x * 256 + threadIdx.x) >> 5;   // global wave = m*K + k
  int lane = threadIdx.x & 31;
  const float* src = cb + (size_t)w * C_ + lane * 8;
  union { __bf16 h[8]; uint4 q; } o;
  float s = 0.0f;
#pragma unroll
  for (int j = 0; j < 8; j++) { float v = src[j]; o.h[j] = (__bf16)v; s += v * v; }
  *(uint4*)(cbb + (size_t)w * C_ + lane * 8) = o.q;
#pragma unroll
  for (int off = 1; off < 32; off <<= 1) s += __shfl_xor(s, off, 32);
  if (lane == 0) cnorm[w] = s;
}

// ---------------------------------------------------------------------------
// Kernel 2: transpose Wp -> WpT[m][c][d], Wo -> WoT[m][blk][c], hi/lo bf16 split
// ---------------------------------------------------------------------------
__global__ __launch_bounds__(256) void prep_w_kernel(const float* __restrict__ Wp,
                                                     const float* __restrict__ Wo,
                                                     __bf16* __restrict__ wpt_hi,
                                                     __bf16* __restrict__ wpt_lo,
                                                     __bf16* __restrict__ wot_hi,
                                                     __bf16* __restrict__ wot_lo) {
  int t   = blockIdx.x * 256 + threadIdx.x;   // 0 .. M*C*BLK-1
  int m   = t / (C_ * BLK_);
  int rem = t % (C_ * BLK_);
  int o   = rem / BLK_;   // output row  (c for WpT, blk for WoT)
  int i   = rem % BLK_;   // output col  (d for WpT, c   for WoT)

  float wp = Wp[((size_t)m * BLK_ + i) * C_ + o];     // Wp[m][d=i][c=o]
  __bf16 h = (__bf16)wp;
  wpt_hi[t] = h; wpt_lo[t] = (__bf16)(wp - (float)h);

  float wo = Wo[((size_t)m * C_ + i) * BLK_ + o];     // Wo[m][c=i][blk=o]
  __bf16 g = (__bf16)wo;
  wot_hi[t] = g; wot_lo[t] = (__bf16)(wo - (float)g);
}

// ---------------------------------------------------------------------------
// A-fragment build from f32 source (split into hi + residual bf16)
// ---------------------------------------------------------------------------
__device__ __forceinline__ void build_A_split(const float* arow, int half,
                                              Frag16 Ah[8], Frag16 Al[8]) {
#pragma unroll
  for (int ks = 0; ks < 8; ks++) {
#pragma unroll
    for (int part = 0; part < 2; part++) {
      const float4* p = (const float4*)(arow + ks * 32 + half * 16 + part * 8);
      float4 f0 = p[0], f1 = p[1];
      float fv[8] = { f0.x, f0.y, f0.z, f0.w, f1.x, f1.y, f1.z, f1.w };
#pragma unroll
      for (int j = 0; j < 4; j++) {
        float a = fv[2 * j], b = fv[2 * j + 1];
        __bf16 ah = (__bf16)a, bh = (__bf16)b;
        Ah[ks].u[part * 4 + j] = packh(ah, bh);
        Al[ks].u[part * 4 + j] = packh((__bf16)(a - (float)ah), (__bf16)(b - (float)bh));
      }
    }
  }
}

// ---------------------------------------------------------------------------
// Kernel 3: x = blocks @ Wp + bp   (split-bf16, 3 WMMAs / K-step)
// block = 256 thr = 8 waves; block covers (m, 128 rows); wave -> 16 rows
// ---------------------------------------------------------------------------
__global__ __launch_bounds__(256) void gemm1_kernel(const float* __restrict__ slots,
                                                    const __bf16* __restrict__ wpt_hi,
                                                    const __bf16* __restrict__ wpt_lo,
                                                    const float* __restrict__ bp,
                                                    float* __restrict__ x_f32,
                                                    __bf16* __restrict__ x_bf16) {
  int m    = blockIdx.x / (BN_ / 128);
  int rb   = (blockIdx.x % (BN_ / 128)) * 128;
  int wave = threadIdx.x >> 5, lane = threadIdx.x & 31;
  int r0   = rb + wave * 16;
  int row  = r0 + (lane & 15);
  int half = lane >> 4;

  const float* arow = slots + (size_t)row * D_ + m * BLK_;   // blocks[row][m][:]
  Frag16 Ah[8], Al[8];
  build_A_split(arow, half, Ah, Al);

  const __bf16* wh = wpt_hi + (size_t)m * C_ * BLK_;
  const __bf16* wl = wpt_lo + (size_t)m * C_ * BLK_;

  for (int nt = 0; nt < 16; nt++) {
    int n0 = nt * 16;
    int col = n0 + (lane & 15);
    FragAcc acc;
#pragma unroll
    for (int i = 0; i < 8; i++) acc.f[i] = 0.0f;
#pragma unroll
    for (int ks = 0; ks < 8; ks++) {
      Frag16 Bh, Bl;
      const uint4* ph = (const uint4*)(wh + (size_t)col * BLK_ + ks * 32 + half * 16);
      const uint4* pl = (const uint4*)(wl + (size_t)col * BLK_ + ks * 32 + half * 16);
      Bh.q[0] = ph[0]; Bh.q[1] = ph[1];
      Bl.q[0] = pl[0]; Bl.q[1] = pl[1];
      acc.v = wmma_bf16(Ah[ks], Bh, acc.v);
      acc.v = wmma_bf16(Ah[ks], Bl, acc.v);
      acc.v = wmma_bf16(Al[ks], Bh, acc.v);
    }
    float bias = bp[m * C_ + col];
#pragma unroll
    for (int r = 0; r < 8; r++) {
      int orow = r0 + r + 8 * half;
      float v = acc.f[r] + bias;
      size_t off = ((size_t)m * BN_ + orow) * C_ + col;
      x_f32[off]  = v;
      x_bf16[off] = (__bf16)v;
    }
  }
}

// ---------------------------------------------------------------------------
// Kernel 4: nearest-codebook search.  argmin_k (||c_k||^2 - 2 x.c_k)
// block covers (m, 128 rows); 32-code tiles double-buffered in LDS via
// GLOBAL_LOAD_ASYNC_TO_LDS_B128; prefetch overlaps the 16 WMMAs per tile.
// ---------------------------------------------------------------------------
__global__ __launch_bounds__(256) void dist_kernel(const __bf16* __restrict__ xb,
                                                   const __bf16* __restrict__ cb,
                                                   const float* __restrict__ cnorm,
                                                   int* __restrict__ idx_out) {
  __shared__ __bf16 tile[2][32 * C_];                  // 2 x 16 KB
  int m    = blockIdx.x / (BN_ / 128);
  int rb   = (blockIdx.x % (BN_ / 128)) * 128;
  int wave = threadIdx.x >> 5, lane = threadIdx.x & 31;
  int r0   = rb + wave * 16;
  int row  = r0 + (lane & 15);
  int half = lane >> 4;

  // A fragments for this wave's 16 rows, all 8 K-steps (x_bf16 channel-contig)
  Frag16 A[8];
  const __bf16* xrow = xb + ((size_t)m * BN_ + row) * C_;
#pragma unroll
  for (int ks = 0; ks < 8; ks++) {
    const uint4* p0 = (const uint4*)(xrow + ks * 32 + half * 16);
    const uint4* p1 = (const uint4*)(xrow + ks * 32 + half * 16 + 8);
    A[ks].q[0] = *p0; A[ks].q[1] = *p1;
  }

  float bestv[8]; int besti[8];
#pragma unroll
  for (int r = 0; r < 8; r++) { bestv[r] = 3.402823466e38f; besti[r] = 0; }

  const __bf16* cbm = cb + (size_t)m * K_ * C_;
  const unsigned lds_base = (unsigned)(uintptr_t)&tile[0][0];

  // per-thread async stage: 64 B of the 16 KB tile (4 x B128)
  auto prefetch_tile = [&](int k0, int buf) {
    unsigned dst = lds_base + (unsigned)buf * (32 * C_ * 2) + threadIdx.x * 64u;
    const __bf16* src = cbm + (size_t)k0 * C_ + threadIdx.x * 32;
#pragma unroll
    for (int j = 0; j < 4; j++)
      async_copy_b128(dst + j * 16u, src + j * 8);
  };

  prefetch_tile(0, 0);

  for (int k0 = 0; k0 < K_; k0 += 32) {
    int buf = (k0 >> 5) & 1;
    wait_asynccnt0();          // own async stores to LDS complete
    __syncthreads();           // everyone's complete; prev buffer fully consumed
    if (k0 + 32 < K_) prefetch_tile(k0 + 32, buf ^ 1);

    const __bf16* t = &tile[buf][0];
#pragma unroll
    for (int sub = 0; sub < 2; sub++) {
      FragAcc acc;
#pragma unroll
      for (int i = 0; i < 8; i++) acc.f[i] = 0.0f;
#pragma unroll
      for (int ks = 0; ks < 8; ks++) {
        Frag16 Bf;
        const uint4* pb = (const uint4*)(t + (sub * 16 + (lane & 15)) * C_ +
                                         ks * 32 + half * 16);
        Bf.q[0] = pb[0]; Bf.q[1] = pb[1];
        acc.v = wmma_bf16(A[ks], Bf, acc.v);
      }
      int   ki = k0 + sub * 16 + (lane & 15);
      float cn = cnorm[m * K_ + ki];
#pragma unroll
      for (int r = 0; r < 8; r++) {
        float v = cn - 2.0f * acc.f[r];
        if (v < bestv[r]) { bestv[r] = v; besti[r] = ki; }
      }
    }
  }

  // reduce across the 16 lanes that share each row (xor 1,2,4,8 stays in half)
#pragma unroll
  for (int r = 0; r < 8; r++) {
    float v = bestv[r]; int i = besti[r];
#pragma unroll
    for (int off = 1; off < 16; off <<= 1) {
      float ov = __shfl_xor(v, off, 32);
      int   oi = __shfl_xor(i, off, 32);
      if (ov < v || (ov == v && oi < i)) { v = ov; i = oi; }
    }
    if ((lane & 15) == 0) {
      int orow = r0 + r + 8 * half;
      idx_out[(size_t)orow * M_ + m] = i;
    }
  }
}

// ---------------------------------------------------------------------------
// Kernel 5: loss = 1.25 * mean((q - x)^2)
// ---------------------------------------------------------------------------
__global__ __launch_bounds__(256) void loss_kernel(const float* __restrict__ x,
                                                   const float* __restrict__ cb,
                                                   const int* __restrict__ idx,
                                                   float* __restrict__ loss) {
  const size_t total = (size_t)M_ * BN_ * C_;
  float acc = 0.0f;
  for (size_t e = (size_t)blockIdx.x * 256 + threadIdx.x; e < total;
       e += (size_t)gridDim.x * 256) {
    int m     = (int)(e / ((size_t)BN_ * C_));
    size_t rm = e % ((size_t)BN_ * C_);
    int row   = (int)(rm / C_);
    int c     = (int)(rm % C_);
    int k     = idx[(size_t)row * M_ + m];
    float q   = cb[((size_t)m * K_ + k) * C_ + c];
    float d   = q - x[e];                   // x laid out [m][row][c] == e
    acc += d * d;
  }
  __shared__ float red[8];
  int lane = threadIdx.x & 31, wave = threadIdx.x >> 5;
#pragma unroll
  for (int off = 1; off < 32; off <<= 1) acc += __shfl_xor(acc, off, 32);
  if (lane == 0) red[wave] = acc;
  __syncthreads();
  if (threadIdx.x == 0) {
    float s = 0.0f;
#pragma unroll
    for (int i = 0; i < 8; i++) s += red[i];
    atomicAdd(loss, s * (1.25f / (float)total));
  }
}

// ---------------------------------------------------------------------------
// Kernel 6: out = q @ Wo + bo  (q rows gathered from codebooks by idx)
// ---------------------------------------------------------------------------
__global__ __launch_bounds__(256) void gemm3_kernel(const float* __restrict__ cb,
                                                    const int* __restrict__ idx,
                                                    const __bf16* __restrict__ wot_hi,
                                                    const __bf16* __restrict__ wot_lo,
                                                    const float* __restrict__ bo,
                                                    float* __restrict__ out) {
  int m    = blockIdx.x / (BN_ / 128);
  int rb   = (blockIdx.x % (BN_ / 128)) * 128;
  int wave = threadIdx.x >> 5, lane = threadIdx.x & 31;
  int r0   = rb + wave * 16;
  int row  = r0 + (lane & 15);
  int half = lane >> 4;

  int kidx = idx[(size_t)row * M_ + m];
  const float* arow = cb + ((size_t)m * K_ + kidx) * C_;   // q row (f32)
  Frag16 Ah[8], Al[8];
  build_A_split(arow, half, Ah, Al);

  const __bf16* wh = wot_hi + (size_t)m * BLK_ * C_;
  const __bf16* wl = wot_lo + (size_t)m * BLK_ * C_;

  for (int nt = 0; nt < 16; nt++) {
    int n0 = nt * 16;
    int col = n0 + (lane & 15);
    FragAcc acc;
#pragma unroll
    for (int i = 0; i < 8; i++) acc.f[i] = 0.0f;
#pragma unroll
    for (int ks = 0; ks < 8; ks++) {
      Frag16 Bh, Bl;
      const uint4* ph = (const uint4*)(wh + (size_t)col * C_ + ks * 32 + half * 16);
      const uint4* pl = (const uint4*)(wl + (size_t)col * C_ + ks * 32 + half * 16);
      Bh.q[0] = ph[0]; Bh.q[1] = ph[1];
      Bl.q[0] = pl[0]; Bl.q[1] = pl[1];
      acc.v = wmma_bf16(Ah[ks], Bh, acc.v);
      acc.v = wmma_bf16(Ah[ks], Bl, acc.v);
      acc.v = wmma_bf16(Al[ks], Bh, acc.v);
    }
    float bias = bo[m * BLK_ + col];
#pragma unroll
    for (int r = 0; r < 8; r++) {
      int orow = r0 + r + 8 * half;
      out[(size_t)orow * D_ + m * BLK_ + col] = acc.f[r] + bias;
    }
  }
}

// ---------------------------------------------------------------------------
// Host-side launcher
// ---------------------------------------------------------------------------
extern "C" void kernel_launch(void* const* d_in, const int* in_sizes, int n_in,
                              void* d_out, int out_size, void* d_ws, size_t ws_size,
                              hipStream_t stream) {
  (void)in_sizes; (void)n_in; (void)out_size; (void)ws_size;
  const float* slots = (const float*)d_in[0];
  const float* Wp    = (const float*)d_in[1];
  const float* bp    = (const float*)d_in[2];
  const float* cbks  = (const float*)d_in[3];
  const float* Wo    = (const float*)d_in[4];
  const float* bo    = (const float*)d_in[5];

  float* out  = (float*)d_out;
  float* loss = out + (size_t)BN_ * D_;            // element 16777216
  int*   idx  = (int*)(out + (size_t)BN_ * D_ + 1);

  char* ws = (char*)d_ws;
  float*  x_f32  = (float*) (ws);                   //  67,108,864 B
  __bf16* x_bf16 = (__bf16*)(ws +  67108864);       //  33,554,432 B
  __bf16* cb_bf  = (__bf16*)(ws + 100663296);       //  16,777,216 B
  float*  cnorm  = (float*) (ws + 117440512);       //     131,072 B
  __bf16* wpt_hi = (__bf16*)(ws + 117571584);       //   1,048,576 B
  __bf16* wpt_lo = (__bf16*)(ws + 118620160);
  __bf16* wot_hi = (__bf16*)(ws + 119668736);
  __bf16* wot_lo = (__bf16*)(ws + 120717312);       // end: 121,765,888 B

  zero_loss_kernel<<<1, 32, 0, stream>>>(loss);
  prep_cb_kernel<<<(M_ * K_) / 8, 256, 0, stream>>>(cbks, cb_bf, cnorm);
  prep_w_kernel<<<(M_ * C_ * BLK_) / 256, 256, 0, stream>>>(Wp, Wo, wpt_hi, wpt_lo,
                                                            wot_hi, wot_lo);
  gemm1_kernel<<<M_ * (BN_ / 128), 256, 0, stream>>>(slots, wpt_hi, wpt_lo, bp,
                                                     x_f32, x_bf16);
  dist_kernel<<<M_ * (BN_ / 128), 256, 0, stream>>>(x_bf16, cb_bf, cnorm, idx);
  loss_kernel<<<1024, 256, 0, stream>>>(x_f32, cbks, idx, loss);
  gemm3_kernel<<<M_ * (BN_ / 128), 256, 0, stream>>>(cbks, idx, wot_hi, wot_lo, bo, out);
}
